// SpatioTemporalGNN_24893630447901
// MI455X (gfx1250) — matrix operand order repaired and checked by reading, not compiled, not run on hardware
//
#include <hip/hip_runtime.h>
#include <hip/hip_bf16.h>
#include <math.h>

typedef __attribute__((ext_vector_type(16))) _Float16 v16h;
typedef __attribute__((ext_vector_type(8)))  float    v8f;
typedef __attribute__((ext_vector_type(4)))  float    f4;
typedef __attribute__((ext_vector_type(4)))  _Float16 h4;

#define BB 4
#define NN 512
#define TT 8
#define FIN 32
#define DD 128
#define DINNER 256
#define DSTATE 16
#define DTRANK 16
#define NTOK 2048          // B*N
#define MROWS 16384        // B*N*T
#define NHEADS 4
#define DHD 32
#define EHID 64
#define KTOP 10
#define NEDGE 20480        // NTOK*KTOP

__device__ __forceinline__ float geluf(float x){ return 0.5f*x*(1.0f+erff(x*0.70710678118654752f)); }
__device__ __forceinline__ float softplusf(float x){ return x>0.f ? x+log1pf(expf(-x)) : log1pf(expf(x)); }
__device__ __forceinline__ float siluf(float x){ return x/(1.0f+expf(-x)); }
__device__ __forceinline__ float eluf(float x){ return x>0.f ? x : (expf(x)-1.0f); }
__device__ __forceinline__ unsigned encf(float f){ unsigned u=__float_as_uint(f); return (u&0x80000000u)?~u:(u|0x80000000u); }
__device__ __forceinline__ float decf(unsigned u){ return (u&0x80000000u)?__uint_as_float(u&0x7fffffffu):__uint_as_float(~u); }

// ---------------------------------------------------------------------------
// f32 GEMM via v_wmma_f32_16x16x32_f16, f32->f16 staged through LDS.
// C[M,N] = act( A[M,K] @ (TRB ? B[N,K]^T : B[K,N]) + bias + resid )
// Block tile 64x64 (4 waves, each 2x2 WMMA tiles), K-step 32.
// FULL: M%64==0 && N%64==0 && K%32==0 -> branch-free staging with
//       ping-pong LDS double buffering (one barrier per K-step).
// ---------------------------------------------------------------------------
template<bool FULL, bool TRB>
__global__ __launch_bounds__(128) void gemm_wmma(
    const float* __restrict__ A, const float* __restrict__ B,
    const float* __restrict__ bias, const float* __restrict__ resid,
    float* __restrict__ C,
    int M, int N, int K, int lda, int ldb, int ldc, int act)
{
  __shared__ _Float16 As[2][64][40];
  __shared__ _Float16 Bs[2][64][40];   // [n][k]
  const int tid  = threadIdx.x;
  const int lane = tid & 31;
  const int wave = tid >> 5;
  const int m0 = blockIdx.y * 64;
  const int n0 = blockIdx.x * 64;
  const int wqm = (wave >> 1) << 5;
  const int wqn = (wave & 1) << 5;

  v8f acc[2][2];
  #pragma unroll
  for (int mi = 0; mi < 2; ++mi)
    #pragma unroll
    for (int ni = 0; ni < 2; ++ni)
      acc[mi][ni] = (v8f){0.f,0.f,0.f,0.f,0.f,0.f,0.f,0.f};

  const int r  = tid >> 2;          // 0..31
  const int cb = (tid & 3) << 3;    // 0,8,16,24

  // fragment build + 4 WMMA from a given LDS buffer
  auto mma_step = [&](const _Float16 (*Ap)[40], const _Float16 (*Bp)[40]) {
    const int half = lane >> 4;
    const int lm2 = lane & 15;
    const int kh = half << 3;       // A frag: K kh..kh+7 and 16+kh..
    const int ks = half << 4;       // B frag: same K order as A frag
    v16h a0, a1, b0, b1;
    #pragma unroll
    for (int e = 0; e < 8; ++e) {
      a0[e]     = Ap[wqm + lm2][kh + e];
      a0[e + 8] = Ap[wqm + lm2][16 + kh + e];
      a1[e]     = Ap[wqm + 16 + lm2][kh + e];
      a1[e + 8] = Ap[wqm + 16 + lm2][16 + kh + e];
    }
    #pragma unroll
    for (int e = 0; e < 16; ++e) {
      b0[e] = Bp[wqn + lm2][kh + (e & 7) + ((e >> 3) << 4)];
      b1[e] = Bp[wqn + 16 + lm2][kh + (e & 7) + ((e >> 3) << 4)];
    }
    acc[0][0] = __builtin_amdgcn_wmma_f32_16x16x32_f16(false, a0, false, b0, (short)0, acc[0][0], false, false);
    acc[0][1] = __builtin_amdgcn_wmma_f32_16x16x32_f16(false, a0, false, b1, (short)0, acc[0][1], false, false);
    acc[1][0] = __builtin_amdgcn_wmma_f32_16x16x32_f16(false, a1, false, b0, (short)0, acc[1][0], false, false);
    acc[1][1] = __builtin_amdgcn_wmma_f32_16x16x32_f16(false, a1, false, b1, (short)0, acc[1][1], false, false);
  };

  if (FULL) {
    f4 ar0, ar1, ar2, ar3, br0, br1, br2, br3;
    auto gload = [&](int k0) {
      const float* ap = A + (long long)(m0 + r) * lda + (k0 + cb);
      ar0 = *(const f4*)ap;              ar1 = *(const f4*)(ap + 4);
      ar2 = *(const f4*)(ap + 32 * lda); ar3 = *(const f4*)(ap + 32 * lda + 4);
      if (TRB) {
        const float* bp = B + (long long)(n0 + r) * ldb + (k0 + cb);
        br0 = *(const f4*)bp;              br1 = *(const f4*)(bp + 4);
        br2 = *(const f4*)(bp + 32 * ldb); br3 = *(const f4*)(bp + 32 * ldb + 4);
      } else {
        const float* bp = B + (long long)(k0 + r) * ldb + n0;
        br0 = *(const f4*)(bp + cb);      br1 = *(const f4*)(bp + cb + 4);
        br2 = *(const f4*)(bp + cb + 32); br3 = *(const f4*)(bp + cb + 36);
      }
      if (k0 + 32 < K) __builtin_prefetch(ap + 32, 0, 1);
    };
    gload(0);
    int p = 0;
    for (int k0 = 0; k0 < K; k0 += 32) {
      // commit staged registers into LDS buffer p (waits on their loads)
      *(h4*)&As[p][r][cb]          = __builtin_convertvector(ar0, h4);
      *(h4*)&As[p][r][cb + 4]      = __builtin_convertvector(ar1, h4);
      *(h4*)&As[p][r + 32][cb]     = __builtin_convertvector(ar2, h4);
      *(h4*)&As[p][r + 32][cb + 4] = __builtin_convertvector(ar3, h4);
      if (TRB) {
        *(h4*)&Bs[p][r][cb]          = __builtin_convertvector(br0, h4);
        *(h4*)&Bs[p][r][cb + 4]      = __builtin_convertvector(br1, h4);
        *(h4*)&Bs[p][r + 32][cb]     = __builtin_convertvector(br2, h4);
        *(h4*)&Bs[p][r + 32][cb + 4] = __builtin_convertvector(br3, h4);
      } else {
        #pragma unroll
        for (int e = 0; e < 4; ++e) {
          Bs[p][cb + e][r]      = (_Float16)br0[e];
          Bs[p][cb + 4 + e][r]  = (_Float16)br1[e];
          Bs[p][cb + 32 + e][r] = (_Float16)br2[e];
          Bs[p][cb + 36 + e][r] = (_Float16)br3[e];
        }
      }
      __syncthreads();                     // buffer p visible to all waves
      if (k0 + 32 < K) gload(k0 + 32);     // next tile in flight during WMMA
      mma_step(As[p], Bs[p]);
      p ^= 1;                              // writers switch buffers; prior
                                           // readers drained before the barrier
    }
  } else {
    // edge tiles: branchless clamp-index loads + value select (no exec waves)
    for (int k0 = 0; k0 < K; k0 += 32) {
      #pragma unroll
      for (int rr = 0; rr < 2; ++rr) {
        const int row = r + (rr << 5);
        const int gm = m0 + row; const bool mok = gm < M;
        const int am = mok ? gm : 0;
        const int gn = n0 + row; const bool nok = gn < N;
        const int an = nok ? gn : 0;
        #pragma unroll
        for (int e = 0; e < 8; ++e) {
          const int gk = k0 + cb + e; const bool kok = gk < K;
          const int ak = kok ? gk : 0;
          float av = A[(long long)am * lda + ak];
          As[0][row][cb + e] = (_Float16)((mok && kok) ? av : 0.f);
          float bv2 = TRB ? B[(long long)an * ldb + ak] : B[(long long)ak * ldb + an];
          Bs[0][row][cb + e] = (_Float16)((nok && kok) ? bv2 : 0.f);
        }
      }
      __syncthreads();
      mma_step(As[0], Bs[0]);
      __syncthreads();
    }
  }

  const int lm = lane & 15;
  const int mb = (lane >> 4) << 3;
  #pragma unroll
  for (int mi = 0; mi < 2; ++mi) {
    #pragma unroll
    for (int ni = 0; ni < 2; ++ni) {
      int gn = n0 + wqn + (ni << 4) + lm;
      int gmBase = m0 + wqm + (mi << 4) + mb;
      if (FULL || gn < N) {
        float bv = bias ? bias[gn] : 0.f;
        #pragma unroll
        for (int v = 0; v < 8; ++v) {
          int gm = gmBase + v;
          if (FULL || gm < M) {
            float xv = acc[mi][ni][v] + bv;
            if (resid) xv += resid[(long long)gm * ldc + gn];
            if (act == 1) xv = geluf(xv);
            else if (act == 2) xv = softplusf(xv);
            C[(long long)gm * ldc + gn] = xv;
          }
        }
      }
    }
  }
}

// LayerNorm over 128 cols; Y = LN(X (+R)) * g + b. One wave per row, f4 per lane.
__global__ __launch_bounds__(128) void ln128(const float* __restrict__ X,
    const float* __restrict__ R, const float* __restrict__ g,
    const float* __restrict__ bta, float* __restrict__ Y, int rows)
{
  int row = blockIdx.x * 4 + (threadIdx.x >> 5);
  int lane = threadIdx.x & 31;
  if (row >= rows) return;
  int c0 = lane << 2;
  f4 xv = *(const f4*)(X + (long long)row * 128 + c0);
  if (R) {
    f4 rv = *(const f4*)(R + (long long)row * 128 + c0);
    xv += rv;
  }
  float s = xv[0] + xv[1] + xv[2] + xv[3];
  #pragma unroll
  for (int o = 16; o > 0; o >>= 1) s += __shfl_xor(s, o, 32);
  float mean = s * (1.0f / 128.0f);
  f4 d = xv - mean;
  float var = d[0]*d[0] + d[1]*d[1] + d[2]*d[2] + d[3]*d[3];
  #pragma unroll
  for (int o = 16; o > 0; o >>= 1) var += __shfl_xor(var, o, 32);
  float inv = rsqrtf(var * (1.0f / 128.0f) + 1e-5f);
  f4 gv = *(const f4*)(g + c0);
  f4 bv = *(const f4*)(bta + c0);
  f4 out = d * inv * gv + bv;
  *(f4*)(Y + (long long)row * 128 + c0) = out;
}

// depthwise causal conv (D_CONV=2) + silu on xz[:, :256]
__global__ void conv_silu(const float* __restrict__ xz, const float* __restrict__ w,
                          const float* __restrict__ b, float* __restrict__ xc)
{
  int t0 = blockIdx.x * 256 + threadIdx.x;
  if (t0 >= MROWS * DINNER) return;
  int c = t0 % DINNER;
  long long r = t0 / DINNER;          // row = token*T + t
  int t = (int)(r % TT);
  float cur  = xz[r * 512 + c];
  float prev = (t > 0) ? xz[(r - 1) * 512 + c] : 0.f;
  xc[r * DINNER + c] = siluf(prev * w[c * 2 + 0] + cur * w[c * 2 + 1] + b[c]);
}

// selective scan; emits only t = T-1, gated by silu(z)
__global__ void ssm_scan(const float* __restrict__ xz, const float* __restrict__ xc,
    const float* __restrict__ dt, const float* __restrict__ xp,
    const float* __restrict__ Alog, const float* __restrict__ Dp,
    float* __restrict__ ylast)
{
  int t0 = blockIdx.x * 256 + threadIdx.x;
  if (t0 >= NTOK * DINNER) return;
  int i = t0 >> 8;
  int c = t0 & 255;
  float A[DSTATE];
  #pragma unroll
  for (int s = 0; s < DSTATE; ++s) A[s] = -expf(Alog[c * DSTATE + s]);
  float hs[DSTATE];
  #pragma unroll
  for (int s = 0; s < DSTATE; ++s) hs[s] = 0.f;
  float y = 0.f;
  for (int t = 0; t < TT; ++t) {
    long long r = (long long)i * TT + t;
    float dtv = dt[r * DINNER + c];
    float xv  = xc[r * DINNER + c];
    const float* Bp = xp + r * 48 + DTRANK;
    f4 Bv[4], Cv[4];
    #pragma unroll
    for (int q = 0; q < 4; ++q) { Bv[q] = *(const f4*)(Bp + 4*q); Cv[q] = *(const f4*)(Bp + DSTATE + 4*q); }
    y = 0.f;
    #pragma unroll
    for (int s = 0; s < DSTATE; ++s) {
      hs[s] = expf(dtv * A[s]) * hs[s] + dtv * Bv[s >> 2][s & 3] * xv;
      y += hs[s] * Cv[s >> 2][s & 3];
    }
  }
  long long rl = (long long)i * TT + (TT - 1);
  float yf = y + xc[rl * DINNER + c] * Dp[c];
  float z  = xz[rl * 512 + DINNER + c];
  ylast[(long long)i * DINNER + c] = yf * siluf(z);
}

__global__ void copy_last(const float* __restrict__ h_tok, float* __restrict__ rlast)
{
  int t = blockIdx.x * 256 + threadIdx.x;
  if (t >= NTOK * 32) return;
  int i = t >> 5, d = (t & 31) << 2;
  *(f4*)(rlast + (long long)i * 128 + d) =
      *(const f4*)(h_tok + ((long long)i * TT + (TT - 1)) * 128 + d);
}

__global__ void fuse_build(const float* __restrict__ henc, const float* __restrict__ emb,
                           const int* __restrict__ ids, float* __restrict__ fused)
{
  int t = blockIdx.x * 256 + threadIdx.x;
  if (t >= NTOK * 160) return;
  int i = t / 160, d = t % 160;
  fused[t] = (d < 128) ? henc[i * 128 + d] : emb[ids[i] * 32 + (d - 128)];
}

__global__ __launch_bounds__(128) void softmax512(float* __restrict__ S, float scale, int rows)
{
  int row = blockIdx.x * 4 + (threadIdx.x >> 5);
  int lane = threadIdx.x & 31;
  if (row >= rows) return;
  float* p = S + (long long)row * 512 + (lane << 4);
  f4 v[4]; float mx = -INFINITY;
  #pragma unroll
  for (int i = 0; i < 4; ++i) {
    v[i] = *(const f4*)(p + (i << 2)) * scale;
    #pragma unroll
    for (int e = 0; e < 4; ++e) mx = fmaxf(mx, v[i][e]);
  }
  #pragma unroll
  for (int o = 16; o > 0; o >>= 1) mx = fmaxf(mx, __shfl_xor(mx, o, 32));
  float s = 0.f;
  #pragma unroll
  for (int i = 0; i < 4; ++i) {
    #pragma unroll
    for (int e = 0; e < 4; ++e) { v[i][e] = expf(v[i][e] - mx); s += v[i][e]; }
  }
  #pragma unroll
  for (int o = 16; o > 0; o >>= 1) s += __shfl_xor(s, o, 32);
  float inv = 1.0f / s;
  #pragma unroll
  for (int i = 0; i < 4; ++i) *(f4*)(p + (i << 2)) = v[i] * inv;
}

// pairwise edge-score MLP via WMMA: per (b,i,j-tile16) compute
// score[j] = gelu(gelu(si[i]+sj[j]+e1b) @ e2w + e2b) . e3w + e3b ; diag=-inf
__global__ __launch_bounds__(128) void edge_scores(
    const float* __restrict__ si, const float* __restrict__ sj,
    const float* __restrict__ e1b, const float* __restrict__ e2w,
    const float* __restrict__ e2b, const float* __restrict__ e3w,
    const float* __restrict__ e3bp, float* __restrict__ scores)
{
  __shared__ _Float16 Bs[32][72];      // e2w transposed f16: [n][k]
  int tid = threadIdx.x, lane = tid & 31, wave = tid >> 5;
  int b = blockIdx.z, i = blockIdx.y;
  for (int t = tid; t < EHID * 32; t += 128) {
    int k = t >> 5, n = t & 31;
    Bs[n][k] = (_Float16)e2w[k * 32 + n];
  }
  __syncthreads();
  const float* sirow = si + ((long long)b * NN + i) * EHID;
  int j0 = blockIdx.x * 64 + wave * 16;
  int half = lane >> 4, mloc = lane & 15;
  int kh = half << 3, ks = half << 4;
  const float* sjrow = sj + ((long long)b * NN + (j0 + mloc)) * EHID;
  v8f acc0 = {0.f,0.f,0.f,0.f,0.f,0.f,0.f,0.f};
  v8f acc1 = acc0;
  #pragma unroll
  for (int kt = 0; kt < 2; ++kt) {
    const int kb1 = kt * 32 + kh;
    const int kb2 = kt * 32 + 16 + kh;
    f4 s0 = *(const f4*)(sirow + kb1); f4 s1 = *(const f4*)(sirow + kb1 + 4);
    f4 s2 = *(const f4*)(sirow + kb2); f4 s3 = *(const f4*)(sirow + kb2 + 4);
    f4 t0 = *(const f4*)(sjrow + kb1); f4 t1 = *(const f4*)(sjrow + kb1 + 4);
    f4 t2 = *(const f4*)(sjrow + kb2); f4 t3 = *(const f4*)(sjrow + kb2 + 4);
    f4 e0 = *(const f4*)(e1b + kb1);   f4 e1v = *(const f4*)(e1b + kb1 + 4);
    f4 e2v = *(const f4*)(e1b + kb2);  f4 e3v = *(const f4*)(e1b + kb2 + 4);
    v16h af;
    #pragma unroll
    for (int e = 0; e < 4; ++e) {
      af[e]      = (_Float16)geluf(s0[e] + t0[e] + e0[e]);
      af[e + 4]  = (_Float16)geluf(s1[e] + t1[e] + e1v[e]);
      af[e + 8]  = (_Float16)geluf(s2[e] + t2[e] + e2v[e]);
      af[e + 12] = (_Float16)geluf(s3[e] + t3[e] + e3v[e]);
    }
    v16h bf0, bf1;
    int nloc = lane & 15;
    #pragma unroll
    for (int e = 0; e < 16; ++e) {
      bf0[e] = Bs[nloc][kt * 32 + kh + (e & 7) + ((e >> 3) << 4)];
      bf1[e] = Bs[16 + nloc][kt * 32 + kh + (e & 7) + ((e >> 3) << 4)];
    }
    acc0 = __builtin_amdgcn_wmma_f32_16x16x32_f16(false, af, false, bf0, (short)0, acc0, false, false);
    acc1 = __builtin_amdgcn_wmma_f32_16x16x32_f16(false, af, false, bf1, (short)0, acc1, false, false);
  }
  int nA = lane & 15, nB = 16 + nA;
  float e3a = e3w[nA], e3b2 = e3w[nB];
  float ba = e2b[nA], bb2 = e2b[nB];
  float e3bias = e3bp[0];
  float red[8];
  #pragma unroll
  for (int v = 0; v < 8; ++v) {
    float xv = geluf(acc0[v] + ba) * e3a + geluf(acc1[v] + bb2) * e3b2;
    xv += __shfl_xor(xv, 1, 32);
    xv += __shfl_xor(xv, 2, 32);
    xv += __shfl_xor(xv, 4, 32);
    xv += __shfl_xor(xv, 8, 32);
    red[v] = xv;
  }
  if ((lane & 15) == 0) {
    int mb = half << 3;
    #pragma unroll
    for (int v = 0; v < 8; ++v) {
      int j = j0 + mb + v;
      float sc = red[v] + e3bias;
      if (j == i) sc = -INFINITY;
      scores[(((long long)b * NN) + i) * NN + j] = sc;
    }
  }
}

// top-10 + softmax per row of 512 scores; one wave per row
__global__ __launch_bounds__(128) void topk_softmax(const float* __restrict__ scores,
    int* __restrict__ tki, float* __restrict__ tew)
{
  int row = blockIdx.x * 4 + (threadIdx.x >> 5);
  int lane = threadIdx.x & 31;
  if (row >= NTOK) return;
  const float* p = scores + (long long)row * NN;
  float v[16];
  #pragma unroll
  for (int i = 0; i < 16; ++i) v[i] = p[lane + (i << 5)];
  unsigned used = 0;
  float vals[KTOP]; int inds[KTOP];
  for (int k = 0; k < KTOP; ++k) {
    float lm = -INFINITY; int li = 0;
    #pragma unroll
    for (int i = 0; i < 16; ++i)
      if (!((used >> i) & 1u) && v[i] > lm) { lm = v[i]; li = i; }
    float bm = lm; int bl = lane, bi = li;
    for (int o = 16; o > 0; o >>= 1) {
      float om = __shfl_xor(bm, o, 32);
      int ol = __shfl_xor(bl, o, 32);
      int oi = __shfl_xor(bi, o, 32);
      if (om > bm || (om == bm && ol < bl)) { bm = om; bl = ol; bi = oi; }
    }
    if (lane == bl) used |= (1u << bi);
    vals[k] = bm; inds[k] = bl + (bi << 5);
  }
  if (lane == 0) {
    float s = 0.f; float ex[KTOP];
    for (int k = 0; k < KTOP; ++k) { ex[k] = expf(vals[k] - vals[0]); s += ex[k]; }
    float inv = 1.0f / s;
    for (int k = 0; k < KTOP; ++k) { tki[row * KTOP + k] = inds[k]; tew[row * KTOP + k] = ex[k] * inv; }
  }
}

__global__ void gat_init(float* gout, float* denom, unsigned* lmax)
{
  int t = blockIdx.x * 256 + threadIdx.x;
  if (t < NTOK * DD) gout[t] = 0.f;
  if (t < NTOK * NHEADS) { denom[t] = 0.f; lmax[t] = 0u; }
}

// per edge: z = leaky_relu(xl[dst]+xr[src]+ew*we); logits[h]=dot(z_h,att_h)
__global__ __launch_bounds__(128) void gat_logits_k(const float* __restrict__ xl,
    const float* __restrict__ xr, const int* __restrict__ tki, const float* __restrict__ tew,
    const float* __restrict__ we, const float* __restrict__ att,
    float* __restrict__ logits, unsigned* __restrict__ lmax)
{
  int e = blockIdx.x * 4 + (threadIdx.x >> 5);
  int lane = threadIdx.x & 31;
  if (e >= NEDGE) return;
  int row = e / KTOP;
  int b = row / NN;
  int dst = b * NN + tki[e];
  float w = tew[e];
  int d0 = lane << 2;
  f4 xlv = *(const f4*)(xl + (long long)dst * DD + d0);
  f4 xrv = *(const f4*)(xr + (long long)row * DD + d0);
  f4 wev = *(const f4*)(we + d0);
  f4 atv = *(const f4*)(att + d0);
  float zs = 0.f;
  #pragma unroll
  for (int t = 0; t < 4; ++t) {
    float z = xlv[t] + xrv[t] + w * wev[t];
    z = (z > 0.f) ? z : 0.2f * z;
    zs += z * atv[t];       // att flat (4,32): index d == h*32+dh
  }
  zs += __shfl_xor(zs, 1, 32);
  zs += __shfl_xor(zs, 2, 32);
  zs += __shfl_xor(zs, 4, 32);
  if ((lane & 7) == 0) {
    int h = lane >> 3;
    logits[e * NHEADS + h] = zs;
    atomicMax(&lmax[dst * NHEADS + h], encf(zs));
  }
}

__global__ void gat_expsum(const float* __restrict__ logits, const int* __restrict__ tki,
    const unsigned* __restrict__ lmax, float* __restrict__ evals, float* __restrict__ denom)
{
  int t = blockIdx.x * 256 + threadIdx.x;
  if (t >= NEDGE * NHEADS) return;
  int e = t >> 2, h = t & 3;
  int row = e / KTOP;
  int b = row / NN;
  int dst = b * NN + tki[e];
  float m = decf(lmax[dst * NHEADS + h]);
  float ev = expf(logits[t] - m);
  evals[t] = ev;
  atomicAdd(&denom[dst * NHEADS + h], ev);
}

__global__ void gat_scatter(const float* __restrict__ xr, const int* __restrict__ tki,
    const float* __restrict__ evals, const float* __restrict__ denom, float* __restrict__ gout)
{
  int t = blockIdx.x * 256 + threadIdx.x;
  if (t >= NEDGE * DD) return;
  int e = t >> 7, d = t & 127;
  int row = e / KTOP;
  int b = row / NN;
  int dst = b * NN + tki[e];
  int h = d >> 5;
  float a = evals[e * NHEADS + h] / (denom[dst * NHEADS + h] + 1e-16f);
  atomicAdd(&gout[(long long)dst * DD + d], xr[(long long)row * DD + d] * a);
}

__global__ void gat_finish(const float* __restrict__ gout, const float* __restrict__ bias,
                           float* __restrict__ hout)
{
  int t = blockIdx.x * 256 + threadIdx.x;
  if (t >= NTOK * DD) return;
  int d = t & 127;
  hout[t] = eluf(gout[t] + bias[d]);
}

__global__ __launch_bounds__(128) void head_out(const float* __restrict__ hh,
    const float* __restrict__ w, const float* __restrict__ b, float* __restrict__ out)
{
  int row = blockIdx.x * 4 + (threadIdx.x >> 5);
  int lane = threadIdx.x & 31;
  if (row >= NTOK) return;
  int c0 = lane << 2;
  f4 hv = *(const f4*)(hh + (long long)row * DD + c0);
  f4 wv = *(const f4*)(w + c0);
  float s = hv[0]*wv[0] + hv[1]*wv[1] + hv[2]*wv[2] + hv[3]*wv[3];
  #pragma unroll
  for (int o = 16; o > 0; o >>= 1) s += __shfl_xor(s, o, 32);
  if (lane == 0) out[row] = s + b[0];
}

// ---------------------------------------------------------------------------
extern "C" void kernel_launch(void* const* d_in, const int* in_sizes, int n_in,
                              void* d_out, int out_size, void* d_ws, size_t ws_size,
                              hipStream_t stream)
{
  (void)in_sizes; (void)n_in; (void)out_size; (void)ws_size;
  const float* x      = (const float*)d_in[0];
  const int*   ids    = (const int*)d_in[1];
  const float* in_w   = (const float*)d_in[2];
  const float* in_b   = (const float*)d_in[3];
  const float* mbn_g  = (const float*)d_in[4];
  const float* mbn_b  = (const float*)d_in[5];
  const float* inproj = (const float*)d_in[6];
  const float* conv_w = (const float*)d_in[7];
  const float* conv_b = (const float*)d_in[8];
  const float* xproj  = (const float*)d_in[9];
  const float* dt_w   = (const float*)d_in[10];
  const float* dt_b   = (const float*)d_in[11];
  const float* Alog   = (const float*)d_in[12];
  const float* Dp     = (const float*)d_in[13];
  const float* outprj = (const float*)d_in[14];
  const float* enc_g  = (const float*)d_in[15];
  const float* enc_b  = (const float*)d_in[16];
  const float* emb    = (const float*)d_in[17];
  const float* fuse_w = (const float*)d_in[18];
  const float* fuse_b = (const float*)d_in[19];
  const float* e1_w   = (const float*)d_in[52];
  const float* e1_b   = (const float*)d_in[53];
  const float* e2_w   = (const float*)d_in[54];
  const float* e2_b   = (const float*)d_in[55];
  const float* e3_w   = (const float*)d_in[56];
  const float* e3_b   = (const float*)d_in[57];
  const float* h_in_w = (const float*)d_in[68];
  const float* h_in_b = (const float*)d_in[69];
  const float* hout_w = (const float*)d_in[88];
  const float* hout_b = (const float*)d_in[89];

  float* W = (float*)d_ws;
  size_t o = 0;
  auto alloc = [&](size_t n) { float* p = W + o; o += (n + 63) & ~(size_t)63; return p; };

  float* henc = alloc((size_t)NTOK * DD);
  float* hbuf = alloc((size_t)NTOK * DD);
  size_t pool = o;
  // phase A (mamba)
  float* h_tok  = alloc((size_t)MROWS * DD);
  float* hn_tok = alloc((size_t)MROWS * DD);
  float* xzb    = alloc((size_t)MROWS * 512);
  float* xcb    = alloc((size_t)MROWS * DINNER);
  float* xpb    = alloc((size_t)MROWS * 48);
  float* dtb    = alloc((size_t)MROWS * DINNER);
  float* ylast  = alloc((size_t)NTOK * DINNER);
  float* rlast  = alloc((size_t)NTOK * DD);
  float* yob    = alloc((size_t)NTOK * DD);
  // phase B (attention/graph/head) reuses the pool
  o = pool;
  float* fused   = alloc((size_t)NTOK * 160);
  float* hnb     = alloc((size_t)NTOK * DD);
  float* qb      = alloc((size_t)NTOK * DD);
  float* kbuf    = alloc((size_t)NTOK * DD);
  float* vb      = alloc((size_t)NTOK * DD);
  float* obuf    = alloc((size_t)NTOK * DD);
  float* attb    = alloc((size_t)BB * NHEADS * NN * NN);
  float* gbuf    = alloc((size_t)NTOK * 512);
  float* sib     = alloc((size_t)NTOK * EHID);
  float* sjb     = alloc((size_t)NTOK * EHID);
  float* scoresb = alloc((size_t)BB * NN * NN);
  int*   tki     = (int*)alloc(NEDGE);
  float* tew     = alloc(NEDGE);
  float* xlb     = alloc((size_t)NTOK * DD);
  float* xrb     = alloc((size_t)NTOK * DD);
  float* logitsb = alloc((size_t)NEDGE * NHEADS);
  float* evalsb  = alloc((size_t)NEDGE * NHEADS);
  float* denomb  = alloc((size_t)NTOK * NHEADS);
  unsigned* lmaxb = (unsigned*)alloc((size_t)NTOK * NHEADS);
  float* goutb   = alloc((size_t)NTOK * DD);
  float* h2b     = alloc((size_t)NTOK * DD);
  float* hhb     = alloc((size_t)NTOK * DD);
  float* rb      = alloc((size_t)NTOK * DD);
  float* t1b     = alloc((size_t)NTOK * DD);

  auto gemm = [&](const float* A, const float* Bm, const float* bias, const float* resid,
                  float* C, int M, int N, int K, int lda, int ldb, int ldc,
                  int transB, int act) {
    dim3 g((N + 63) / 64, (M + 63) / 64, 1);
    bool full = (M % 64 == 0) && (N % 64 == 0) && (K % 32 == 0);
    if (full) {
      if (transB) gemm_wmma<true, true><<<g, dim3(128,1,1), 0, stream>>>(A, Bm, bias, resid, C, M, N, K, lda, ldb, ldc, act);
      else        gemm_wmma<true, false><<<g, dim3(128,1,1), 0, stream>>>(A, Bm, bias, resid, C, M, N, K, lda, ldb, ldc, act);
    } else {
      if (transB) gemm_wmma<false, true><<<g, dim3(128,1,1), 0, stream>>>(A, Bm, bias, resid, C, M, N, K, lda, ldb, ldc, act);
      else        gemm_wmma<false, false><<<g, dim3(128,1,1), 0, stream>>>(A, Bm, bias, resid, C, M, N, K, lda, ldb, ldc, act);
    }
  };

  // ---- Mamba encoder ----
  gemm(x, in_w, in_b, nullptr, h_tok, MROWS, DD, FIN, FIN, DD, DD, 0, 0);
  ln128<<<dim3((MROWS + 3) / 4), 128, 0, stream>>>(h_tok, nullptr, mbn_g, mbn_b, hn_tok, MROWS);
  gemm(hn_tok, inproj, nullptr, nullptr, xzb, MROWS, 512, DD, DD, 512, 512, 0, 0);
  conv_silu<<<dim3((MROWS * DINNER + 255) / 256), 256, 0, stream>>>(xzb, conv_w, conv_b, xcb);
  gemm(xcb, xproj, nullptr, nullptr, xpb, MROWS, 48, DINNER, DINNER, 48, 48, 0, 0);
  gemm(xpb, dt_w, dt_b, nullptr, dtb, MROWS, DINNER, DTRANK, 48, DINNER, DINNER, 0, 2);
  ssm_scan<<<dim3((NTOK * DINNER + 255) / 256), 256, 0, stream>>>(xzb, xcb, dtb, xpb, Alog, Dp, ylast);
  copy_last<<<dim3((NTOK * 32 + 255) / 256), 256, 0, stream>>>(h_tok, rlast);
  gemm(ylast, outprj, nullptr, rlast, yob, NTOK, DD, DINNER, DINNER, DD, DD, 0, 0);
  ln128<<<dim3(NTOK / 4), 128, 0, stream>>>(yob, nullptr, enc_g, enc_b, henc, NTOK);

  // ---- fuse with stock embedding ----
  fuse_build<<<dim3((NTOK * 160 + 255) / 256), 256, 0, stream>>>(henc, emb, ids, fused);
  gemm(fused, fuse_w, fuse_b, nullptr, hbuf, NTOK, DD, 160, 160, DD, DD, 0, 0);

  // ---- cross attention x2 ----
  const float scale = 0.17677669529663687f; // 1/sqrt(DH)
  for (int l = 0; l < 2; ++l) {
    int base = 20 + l * 16;
    const float* n1g = (const float*)d_in[base + 0]; const float* n1b = (const float*)d_in[base + 1];
    const float* wq  = (const float*)d_in[base + 2]; const float* bq  = (const float*)d_in[base + 3];
    const float* wk  = (const float*)d_in[base + 4]; const float* bk  = (const float*)d_in[base + 5];
    const float* wv  = (const float*)d_in[base + 6]; const float* bv  = (const float*)d_in[base + 7];
    const float* wo  = (const float*)d_in[base + 8]; const float* bo  = (const float*)d_in[base + 9];
    const float* n2g = (const float*)d_in[base + 10]; const float* n2b = (const float*)d_in[base + 11];
    const float* f1w = (const float*)d_in[base + 12]; const float* f1b = (const float*)d_in[base + 13];
    const float* f2w = (const float*)d_in[base + 14]; const float* f2b = (const float*)d_in[base + 15];
    ln128<<<dim3(NTOK / 4), 128, 0, stream>>>(hbuf, nullptr, n1g, n1b, hnb, NTOK);
    gemm(hnb, wq, bq, nullptr, qb,   NTOK, DD, DD, DD, DD, DD, 0, 0);
    gemm(hnb, wk, bk, nullptr, kbuf, NTOK, DD, DD, DD, DD, DD, 0, 0);
    gemm(hnb, wv, bv, nullptr, vb,   NTOK, DD, DD, DD, DD, DD, 0, 0);
    for (int b = 0; b < BB; ++b)
      for (int hd = 0; hd < NHEADS; ++hd) {
        long long boff = (long long)b * NN * DD + hd * DHD;
        float* Sb = attb + ((long long)b * NHEADS + hd) * NN * NN;
        gemm(qb + boff, kbuf + boff, nullptr, nullptr, Sb, NN, NN, DHD, DD, DD, NN, 1, 0);
      }
    softmax512<<<dim3(BB * NHEADS * NN / 4), 128, 0, stream>>>(attb, scale, BB * NHEADS * NN);
    for (int b = 0; b < BB; ++b)
      for (int hd = 0; hd < NHEADS; ++hd) {
        long long boff = (long long)b * NN * DD + hd * DHD;
        float* Sb = attb + ((long long)b * NHEADS + hd) * NN * NN;
        gemm(Sb, vb + boff, nullptr, nullptr, obuf + boff, NN, DHD, NN, NN, DD, DD, 0, 0);
      }
    gemm(obuf, wo, bo, hbuf, hbuf, NTOK, DD, DD, DD, DD, DD, 0, 0);
    ln128<<<dim3(NTOK / 4), 128, 0, stream>>>(hbuf, nullptr, n2g, n2b, hnb, NTOK);
    gemm(hnb, f1w, f1b, nullptr, gbuf, NTOK, 512, DD, DD, 512, 512, 0, 1);
    gemm(gbuf, f2w, f2b, hbuf, hbuf, NTOK, DD, 512, 512, DD, DD, 0, 0);
  }

  // ---- graph build ----
  gemm(hbuf, e1_w, nullptr, nullptr, sib, NTOK, EHID, DD, DD, EHID, EHID, 0, 0);
  gemm(hbuf, e1_w + (size_t)DD * EHID, nullptr, nullptr, sjb, NTOK, EHID, DD, DD, EHID, EHID, 0, 0);
  edge_scores<<<dim3(NN / 64, NN, BB), 128, 0, stream>>>(sib, sjb, e1_b, e2_w, e2_b, e3_w, e3_b, scoresb);
  topk_softmax<<<dim3(NTOK / 4), 128, 0, stream>>>(scoresb, tki, tew);

  // ---- GATv2 x2 ----
  float* hcur = hbuf;
  float* hnext = h2b;
  for (int l = 0; l < 2; ++l) {
    int gb = 58 + l * 5;
    const float* wl    = (const float*)d_in[gb + 0];
    const float* wr    = (const float*)d_in[gb + 1];
    const float* we    = (const float*)d_in[gb + 2];
    const float* gatt  = (const float*)d_in[gb + 3];
    const float* gbias = (const float*)d_in[gb + 4];
    gemm(hcur, wl, nullptr, nullptr, xlb, NTOK, DD, DD, DD, DD, DD, 0, 0);
    gemm(hcur, wr, nullptr, nullptr, xrb, NTOK, DD, DD, DD, DD, DD, 0, 0);
    gat_init<<<dim3((NTOK * DD + 255) / 256), 256, 0, stream>>>(goutb, denomb, lmaxb);
    gat_logits_k<<<dim3((NEDGE + 3) / 4), 128, 0, stream>>>(xlb, xrb, tki, tew, we, gatt, logitsb, lmaxb);
    gat_expsum<<<dim3((NEDGE * NHEADS + 255) / 256), 256, 0, stream>>>(logitsb, tki, lmaxb, evalsb, denomb);
    gat_scatter<<<dim3((NEDGE * DD + 255) / 256), 256, 0, stream>>>(xrb, tki, evalsb, denomb, goutb);
    gat_finish<<<dim3((NTOK * DD + 255) / 256), 256, 0, stream>>>(goutb, gbias, hnext);
    float* tmp = hcur; hcur = hnext; hnext = tmp;
  }

  // ---- head MLP ----
  gemm(hcur, h_in_w, h_in_b, nullptr, hhb, NTOK, DD, DD, DD, DD, DD, 0, 1);
  for (int bk2 = 0; bk2 < 3; ++bk2) {
    int hb = 70 + bk2 * 6;
    const float* w1 = (const float*)d_in[hb + 0]; const float* b1 = (const float*)d_in[hb + 1];
    const float* w2 = (const float*)d_in[hb + 2]; const float* b2 = (const float*)d_in[hb + 3];
    const float* ng = (const float*)d_in[hb + 4]; const float* nb = (const float*)d_in[hb + 5];
    gemm(hhb, w1, b1, nullptr, t1b, NTOK, DD, DD, DD, DD, DD, 0, 1);
    gemm(t1b, w2, b2, nullptr, rb, NTOK, DD, DD, DD, DD, DD, 0, 0);
    ln128<<<dim3(NTOK / 4), 128, 0, stream>>>(rb, hhb, ng, nb, hhb, NTOK);
  }
  head_out<<<dim3(NTOK / 4), 128, 0, stream>>>(hhb, hout_w, hout_b, (float*)d_out);
}
